// NeuralCDE_84189948936255
// MI455X (gfx1250) — matrix operand order repaired and checked by reading, not compiled, not run on hardware
//
#include <hip/hip_runtime.h>
#include <hip/hip_bf16.h>

typedef __attribute__((ext_vector_type(16))) __bf16 v16bf;
typedef __attribute__((ext_vector_type(8)))  float  v8f;

#define B_   128
#define T_   1024
#define D_   32
#define H_   64
#define W_   128
#define C_   33      // D+1
#define O_   2112    // H*C
#define KP0  96      // layer-0 K padded 65 -> 96
#define TR   16      // batch rows per block (WMMA M)
#define NT   256     // threads per block (8 waves)

// ---- LDS layout (bytes) ----
#define OFF_W0   0
#define SZ_W0    (W_*KP0*2)          // 24576
#define OFF_W1   (OFF_W0+SZ_W0)      // 24576
#define SZ_W1    (W_*W_*2)           // 32768
#define OFF_V    (OFF_W1+SZ_W1)      // 57344
#define SZ_V     (TR*O_*4)           // 135168
#define OFF_Y    (OFF_V+SZ_V)        // 192512
#define SZ_Y     (TR*H_*4)           // 4096
#define OFF_YH   (OFF_Y+SZ_Y)
#define OFF_SO   (OFF_YH+SZ_Y)
#define OFF_DX   (OFF_SO+SZ_Y)       // 204800
#define SZ_DX    2304                // 16*33*4 padded
#define OFF_XC   (OFF_DX+SZ_DX)
#define OFF_INP  (OFF_XC+SZ_DX)      // 209408
#define SZ_INP   (TR*KP0*2)          // 3072
#define OFF_Z0   (OFF_INP+SZ_INP)
#define SZ_Z     (TR*W_*2)           // 4096
#define OFF_Z1   (OFF_Z0+SZ_Z)
#define SMEM_TOTAL (OFF_Z1+SZ_Z)     // 220672 B < 320 KB/WGP

// ---------- WMMA operand loaders (16x16x32 bf16, wave32 layouts per ISA 7.12.2) ----------
// A (16xK row-major): lane<16 -> row=lane, K = kb+0..7 & kb+16..23 ; lane>=16 -> row=lane-16, K = kb+8..15 & kb+24..31
__device__ __forceinline__ v16bf load_a16x32(const __hip_bfloat16* base, int kpad, int kb, int lane) {
    const __hip_bfloat16* p = base + (lane & 15) * kpad + kb + ((lane & 16) ? 8 : 0);
    v16bf a;
    ((uint4*)&a)[0] = *reinterpret_cast<const uint4*>(p);
    ((uint4*)&a)[1] = *reinterpret_cast<const uint4*>(p + 16);
    return a;
}
// B from row-major weight W[N][K] (B = W^T): lane<16 -> col n=nb+lane, K=kb+0..7 & kb+16..23 ; lane>=16 -> n=nb+lane-16, K=kb+8..15 & kb+24..31
__device__ __forceinline__ v16bf load_b32x16(const __hip_bfloat16* w, int krow, int nb, int kb, int lane) {
    const __hip_bfloat16* p = w + (size_t)(nb + (lane & 15)) * krow + kb + ((lane & 16) ? 8 : 0);
    v16bf b;
    ((uint4*)&b)[0] = *reinterpret_cast<const uint4*>(p);
    ((uint4*)&b)[1] = *reinterpret_cast<const uint4*>(p + 16);
    return b;
}
__device__ __forceinline__ v8f wmma_bf16(v16bf a, v16bf b, v8f c) {
    return __builtin_amdgcn_wmma_f32_16x16x32_bf16(false, a, false, b, (short)0, c, false, false);
}
// lipswish = 0.909 * x * sigmoid(x); native v_rcp_f32 avoids the IEEE divide expansion
__device__ __forceinline__ float lipswish(float x) {
    return 0.909f * x * __builtin_amdgcn_rcpf(1.0f + __expf(-x));
}
// CDNA5 has a hardware TANH transcendental (TRANS32 class): single v_tanh_f32
__device__ __forceinline__ float fast_tanh(float x) {
    float r;
    asm("v_tanh_f32 %0, %1" : "=v"(r) : "v"(x));
    return r;
}

// ---------- vector field: v = tanh(L2(lipswish(L1(lipswish(L0(inp)))))) ----------
__device__ __forceinline__ void run_vf(
    const __hip_bfloat16* inp_bf, const __hip_bfloat16* w0s, const __hip_bfloat16* w1s,
    const __hip_bfloat16* w2g, __hip_bfloat16* z0_bf, __hip_bfloat16* z1_bf, float* vbuf,
    const float* vb2, float b0bias, float b1bias, int ntiles2, int wave, int lane)
{
    const int nsel = (lane & 16) ? 8 : 0;   // D-tile row half
    const int ncol = lane & 15;             // D-tile column
    // ---- Layer 0: (16x96) x (96x128), wave w owns N-tile w ----
    {
        const int nb = wave * 16;
        v8f acc = {};
        #pragma unroll
        for (int kk = 0; kk < KP0; kk += 32)
            acc = wmma_bf16(load_a16x32(inp_bf, KP0, kk, lane),
                            load_b32x16(w0s, KP0, nb, kk, lane), acc);
        const int n = nb + ncol;
        #pragma unroll
        for (int g = 0; g < 8; ++g)
            z0_bf[(g + nsel) * W_ + n] = __float2bfloat16(lipswish(acc[g] + b0bias));
    }
    __syncthreads();
    // ---- Layer 1: (16x128) x (128x128) ----
    {
        const int nb = wave * 16;
        v8f acc = {};
        #pragma unroll
        for (int kk = 0; kk < W_; kk += 32)
            acc = wmma_bf16(load_a16x32(z0_bf, W_, kk, lane),
                            load_b32x16(w1s, W_, nb, kk, lane), acc);
        const int n = nb + ncol;
        #pragma unroll
        for (int g = 0; g < 8; ++g)
            z1_bf[(g + nsel) * W_ + n] = __float2bfloat16(lipswish(acc[g] + b1bias));
    }
    __syncthreads();
    // ---- Layer 2: (16x128) x (128x2112), 132 N-tiles round-robin over 8 waves ----
    for (int j = 0; j < ntiles2; ++j) {
        const int nb = (wave + 8 * j) * 16;
        if (j + 1 < ntiles2) {  // prefetch next weight tile rows into L2/L0
            const int nbn = (wave + 8 * (j + 1)) * 16;
            __builtin_prefetch(w2g + (size_t)(nbn + ncol) * W_ + ((lane & 16) ? 16 : 0), 0, 1);
        }
        v8f acc = {};
        #pragma unroll
        for (int kk = 0; kk < W_; kk += 32)
            acc = wmma_bf16(load_a16x32(z1_bf, W_, kk, lane),
                            load_b32x16(w2g, W_, nb, kk, lane), acc);
        const int n = nb + ncol;
        const float bias2 = vb2[n];
        #pragma unroll
        for (int g = 0; g < 8; ++g)
            vbuf[(size_t)(g + nsel) * O_ + n] = fast_tanh(acc[g] + bias2);
    }
    __syncthreads();
}

// ---------- one-time fp32 -> bf16 weight conversion (layer-0 K padded to 96) ----------
__global__ void cde_convert(const float* vW0, const float* vW1, const float* vW2,
                            __hip_bfloat16* w0, __hip_bfloat16* w1, __hip_bfloat16* w2) {
    const int i0 = blockIdx.x * blockDim.x + threadIdx.x;
    const int stride = gridDim.x * blockDim.x;
    for (int i = i0; i < W_ * KP0; i += stride) {
        int n = i / KP0, k = i - n * KP0;
        w0[i] = __float2bfloat16(k < (H_ + 1) ? vW0[n * (H_ + 1) + k] : 0.0f);
    }
    for (int i = i0; i < W_ * W_; i += stride) w1[i] = __float2bfloat16(vW1[i]);
    for (int i = i0; i < O_ * W_; i += stride) w2[i] = __float2bfloat16(vW2[i]);
}

// ---------- persistent kernel: one block per 16 batch rows, loops over all 1023 steps ----------
__global__ __launch_bounds__(NT) void cde_main(
    const float* __restrict__ ts, const float* __restrict__ ys,
    const float* __restrict__ iW0, const float* __restrict__ ib0,
    const float* __restrict__ iW1, const float* __restrict__ ib1,
    const float* __restrict__ iW2, const float* __restrict__ ib2,
    const float* __restrict__ vb0, const float* __restrict__ vb1,
    const float* __restrict__ vb2, const float* __restrict__ rW,
    const float* __restrict__ rb,
    const __hip_bfloat16* __restrict__ w0g, const __hip_bfloat16* __restrict__ w1g,
    const __hip_bfloat16* __restrict__ w2g, float* __restrict__ out)
{
    extern __shared__ char smem[];
    __hip_bfloat16* w0s    = (__hip_bfloat16*)(smem + OFF_W0);
    __hip_bfloat16* w1s    = (__hip_bfloat16*)(smem + OFF_W1);
    float*          vbuf   = (float*)(smem + OFF_V);
    float*          y      = (float*)(smem + OFF_Y);
    float*          yhat   = (float*)(smem + OFF_YH);
    float*          s_old  = (float*)(smem + OFF_SO);
    float*          dx     = (float*)(smem + OFF_DX);
    float*          xcur   = (float*)(smem + OFF_XC);
    __hip_bfloat16* inp_bf = (__hip_bfloat16*)(smem + OFF_INP);
    __hip_bfloat16* z0_bf  = (__hip_bfloat16*)(smem + OFF_Z0);
    __hip_bfloat16* z1_bf  = (__hip_bfloat16*)(smem + OFF_Z1);

    const int tid  = threadIdx.x;
    const int wave = tid >> 5, lane = tid & 31;
    const int b0   = blockIdx.x * TR;

    // stage layer-0/1 weights into LDS
    { const unsigned int* s = (const unsigned int*)w0g; unsigned int* d = (unsigned int*)w0s;
      for (int i = tid; i < SZ_W0 / 4; i += NT) d[i] = s[i]; }
    { const unsigned int* s = (const unsigned int*)w1g; unsigned int* d = (unsigned int*)w1s;
      for (int i = tid; i < SZ_W1 / 4; i += NT) d[i] = s[i]; }

    // xcur = xs[:,0] = [ts[0], ys[:,0,:]]
    for (int i = tid; i < TR * C_; i += NT) {
        int r = i / C_, c = i - r * C_;
        xcur[i] = (c == 0) ? ts[0] : ys[(size_t)(b0 + r) * T_ * D_ + (c - 1)];
    }
    __syncthreads();

    // initial MLP -> y0 (scratch za/zb inside vbuf, free at this point)
    float* za = vbuf; float* zb = vbuf + TR * W_;
    for (int i = tid; i < TR * W_; i += NT) {
        int r = i >> 7, n = i & 127;
        float s = ib0[n];
        for (int c = 0; c < C_; ++c) s += xcur[r * C_ + c] * iW0[n * C_ + c];
        za[i] = fmaxf(s, 0.0f);
    }
    __syncthreads();
    for (int i = tid; i < TR * W_; i += NT) {
        int r = i >> 7, n = i & 127;
        float s = ib1[n];
        for (int k = 0; k < W_; ++k) s += za[r * W_ + k] * iW1[n * W_ + k];
        zb[i] = fmaxf(s, 0.0f);
    }
    __syncthreads();
    for (int i = tid; i < TR * H_; i += NT) {
        int r = i >> 6, h = i & 63;
        float s = ib2[h];
        for (int k = 0; k < W_; ++k) s += zb[r * W_ + k] * iW2[h * W_ + k];
        y[i] = s; yhat[i] = s;
    }
    // zero-pad inp columns 65..95 once (never touched again)
    for (int i = tid; i < TR * (KP0 - (H_ + 1)); i += NT) {
        int r = i / (KP0 - (H_ + 1)), c = (H_ + 1) + i % (KP0 - (H_ + 1));
        inp_bf[r * KP0 + c] = __float2bfloat16(0.0f);
    }
    __syncthreads();

    const int ncol = lane & 15;
    const float b0bias = vb0[wave * 16 + ncol];
    const float b1bias = vb1[wave * 16 + ncol];
    const int ntiles2 = (O_ / 16 - wave + 7) / 8;  // 132 tiles over 8 waves

    // v = vf(ts[0], y0)
    for (int i = tid; i < TR * H_; i += NT) {
        int r = i >> 6, h = i & 63;
        inp_bf[r * KP0 + 1 + h] = __float2bfloat16(y[i]);
    }
    if (tid < TR) inp_bf[tid * KP0] = __float2bfloat16(ts[0]);
    __syncthreads();
    run_vf(inp_bf, w0s, w1s, w2g, z0_bf, z1_bf, vbuf, vb2, b0bias, b1bias, ntiles2, wave, lane);

    // ---- main sequential scan ----
    for (int i = 0; i < T_ - 1; ++i) {
        const float t1 = ts[i + 1];
        // A: dx = xs[:,i+1] - xs[:,i]
        for (int k = tid; k < TR * C_; k += NT) {
            int r = k / C_, c = k - r * C_;
            float val = (c == 0) ? t1
                : ys[(size_t)(b0 + r) * T_ * D_ + (size_t)(i + 1) * D_ + (c - 1)];
            dx[k] = val - xcur[k];
            xcur[k] = val;
        }
        __syncthreads();
        // B: s_old = v.dx ; yhat1 = 2y - yhat + s_old ; build vf input
        for (int k = tid; k < TR * H_; k += NT) {
            int r = k >> 6, h = k & 63;
            const float* vr  = vbuf + (size_t)r * O_ + h * C_;
            const float* dxr = dx + r * C_;
            float s = 0.0f;
            #pragma unroll
            for (int c = 0; c < C_; ++c) s += vr[c] * dxr[c];
            s_old[k] = s;
            float yh1 = 2.0f * y[k] - yhat[k] + s;
            yhat[k] = yh1;
            inp_bf[r * KP0 + 1 + h] = __float2bfloat16(yh1);
        }
        if (tid < TR) inp_bf[tid * KP0] = __float2bfloat16(t1);
        __syncthreads();
        // v1 = vf(t1, yhat1)  (overwrites vbuf)
        run_vf(inp_bf, w0s, w1s, w2g, z0_bf, z1_bf, vbuf, vb2, b0bias, b1bias, ntiles2, wave, lane);
        // C: y += 0.5*(s_old + v1.dx)
        for (int k = tid; k < TR * H_; k += NT) {
            int r = k >> 6, h = k & 63;
            const float* vr  = vbuf + (size_t)r * O_ + h * C_;
            const float* dxr = dx + r * C_;
            float s = 0.0f;
            #pragma unroll
            for (int c = 0; c < C_; ++c) s += vr[c] * dxr[c];
            y[k] += 0.5f * (s_old[k] + s);
        }
        __syncthreads();
    }

    // readout: out = yT @ rW.T + rb
    if (tid < TR) {
        float s = rb[0];
        for (int h = 0; h < H_; ++h) s += y[tid * H_ + h] * rW[h];
        out[b0 + tid] = s;
    }
}

extern "C" void kernel_launch(void* const* d_in, const int* in_sizes, int n_in,
                              void* d_out, int out_size, void* d_ws, size_t ws_size,
                              hipStream_t stream) {
    (void)in_sizes; (void)n_in; (void)out_size; (void)ws_size;
    const float* ts  = (const float*)d_in[0];
    const float* ys  = (const float*)d_in[1];
    const float* iW0 = (const float*)d_in[2];
    const float* ib0 = (const float*)d_in[3];
    const float* iW1 = (const float*)d_in[4];
    const float* ib1 = (const float*)d_in[5];
    const float* iW2 = (const float*)d_in[6];
    const float* ib2 = (const float*)d_in[7];
    const float* vW0 = (const float*)d_in[8];
    const float* vb0 = (const float*)d_in[9];
    const float* vW1 = (const float*)d_in[10];
    const float* vb1 = (const float*)d_in[11];
    const float* vW2 = (const float*)d_in[12];
    const float* vb2 = (const float*)d_in[13];
    const float* rW  = (const float*)d_in[14];
    const float* rb  = (const float*)d_in[15];

    char* ws = (char*)d_ws;
    __hip_bfloat16* w0g = (__hip_bfloat16*)(ws);                  // 128x96  bf16
    __hip_bfloat16* w1g = (__hip_bfloat16*)(ws + 24576);          // 128x128 bf16
    __hip_bfloat16* w2g = (__hip_bfloat16*)(ws + 24576 + 32768);  // 2112x128 bf16

    cde_convert<<<512, 256, 0, stream>>>(vW0, vW1, vW2, w0g, w1g, w2g);
    cde_main<<<B_ / TR, NT, SMEM_TOTAL, stream>>>(
        ts, ys, iW0, ib0, iW1, ib1, iW2, ib2,
        vb0, vb1, vb2, rW, rb, w0g, w1g, w2g, (float*)d_out);
}
